// QuantumApproximateOptimization_23579370455599
// MI455X (gfx1250) — matrix operand order repaired and checked by reading, not compiled, not run on hardware
//
#include <hip/hip_runtime.h>
#include <math.h>

typedef __attribute__((ext_vector_type(16))) _Float16 v16h;
typedef __attribute__((ext_vector_type(8)))  float    v8f;

#define NVAR    20
#define NCON    64
#define NLAYER  3
#define NSTATES (1 << 20)
#define TOPK    10

// ---------------------------------------------------------------------------
// Cost kernel: violations = bits(state) @ constraints^T via WMMA f16 (hi+lo
// split for ~f32 accuracy), then cost = sum(relu(v-1)^2) + 0.1*popcount.
// One wave (32 threads) per block; each block handles 256 states (16 groups
// of 16). 4 constraint tiles of 16 -> 8 v_wmma per group.
// ---------------------------------------------------------------------------
__global__ void qaoa_cost_wmma(const float* __restrict__ con,
                               float* __restrict__ costs)
{
    const int lane = threadIdx.x;   // 0..31
    const int h    = lane >> 4;     // lane half
    const int n    = lane & 15;

    // B tiles: B[K=16h+j][N=n] = constraints[16t+n][K], split hi/lo f16.
    v16h Bhi[4], Blo[4];
    for (int t = 0; t < 4; ++t) {
        for (int j = 0; j < 16; ++j) {
            const int K = 16 * h + j;
            const float w = (K < NVAR) ? con[(16 * t + n) * NVAR + K] : 0.0f;
            const _Float16 whi = (_Float16)w;
            Bhi[t][j] = whi;
            Blo[t][j] = (_Float16)(w - (float)whi);
        }
    }

    const int blockBase = blockIdx.x * 256;
    for (int g = 0; g < 16; ++g) {
        const int base = blockBase + g * 16;
        const int sA   = base + n;          // A row M = lane%16 (both halves)

        // A element j: K = 16h+j; value = bit(19-K).
        v16h A;
        #pragma unroll
        for (int j = 0; j < 16; ++j) {
            const int K = 16 * h + j;
            _Float16 bit = (_Float16)0;
            if (K < NVAR) bit = (_Float16)(float)((sA >> (NVAR - 1 - K)) & 1);
            A[j] = bit;
        }

        float P[8];
        #pragma unroll
        for (int v = 0; v < 8; ++v) P[v] = 0.0f;

        #pragma unroll
        for (int t = 0; t < 4; ++t) {
            v8f C = {};
            C = __builtin_amdgcn_wmma_f32_16x16x32_f16(false, A, false, Bhi[t],
                                                       (short)0, C, false, false);
            C = __builtin_amdgcn_wmma_f32_16x16x32_f16(false, A, false, Blo[t],
                                                       (short)0, C, false, false);
            #pragma unroll
            for (int v = 0; v < 8; ++v) {
                float x = C[v] - 1.0f;          // relu(viol - 1)^2
                x = fmaxf(x, 0.0f);
                P[v] += x * x;
            }
        }

        // Reduce over constraint axis (16 lanes of each half).
        #pragma unroll
        for (int v = 0; v < 8; ++v) {
            float s = P[v];
            s += __shfl_xor(s, 1, 32);
            s += __shfl_xor(s, 2, 32);
            s += __shfl_xor(s, 4, 32);
            s += __shfl_xor(s, 8, 32);
            P[v] = s;
        }

        // C layout: VGPR v -> state base+v (half 0) / base+8+v (half 1).
        if (n == 0) {
            #pragma unroll
            for (int v = 0; v < 8; ++v) {
                const int st = base + 8 * h + v;
                costs[st] = P[v] + 0.1f * (float)__popc(st);
            }
        }
    }
}

// ---------------------------------------------------------------------------
// amps *= scale * cos(gamma*cost). Layer 0: amps = 1/1024 (uniform init).
// Previous layer's L2 normalization is folded in via rsqrt(normsq_prev).
// ---------------------------------------------------------------------------
__global__ void qaoa_scale_cos(const float* __restrict__ src,
                               const float* __restrict__ costs,
                               const float* __restrict__ gamma, int layer,
                               const float* __restrict__ normsq_prev,
                               float* __restrict__ dst)
{
    const int i = blockIdx.x * blockDim.x + threadIdx.x;
    const float g = gamma[layer];
    float a;
    if (layer == 0)
        a = 1.0f / 1024.0f;                       // 1/sqrt(2^20)
    else
        a = src[i] * rsqrtf(normsq_prev[0]);
    dst[i] = a * cosf(g * costs[i]);
}

// ---------------------------------------------------------------------------
// Mixer: c[i] = b[i] + 0.1*sin(beta)*sum_{bit<20} b[i ^ (1<<bit)].
// Block handles 1024 contiguous states; bits 0..9 neighbors read from LDS,
// bits 10..19 from global (L2-resident, 4MB array). Per-block norm^2 partial.
// ---------------------------------------------------------------------------
__global__ void qaoa_mix(const float* __restrict__ b, float* __restrict__ c,
                         const float* __restrict__ beta, int layer,
                         float* __restrict__ partials)
{
    __shared__ float tile[1024];
    __shared__ float red[256];
    const int tid  = threadIdx.x;
    const int base = blockIdx.x * 1024;

    #pragma unroll
    for (int q = 0; q < 4; ++q)
        tile[tid + 256 * q] = b[base + tid + 256 * q];
    __syncthreads();

    const float mc = 0.1f * sinf(beta[layer]);
    float local = 0.0f;
    #pragma unroll
    for (int q = 0; q < 4; ++q) {
        const int il = tid + 256 * q;
        const int i  = base + il;
        float ns = 0.0f;
        #pragma unroll
        for (int bit = 0; bit < 10; ++bit) ns += tile[il ^ (1 << bit)];
        #pragma unroll
        for (int bit = 10; bit < 20; ++bit) ns += b[i ^ (1 << bit)];
        const float o = tile[il] + mc * ns;
        c[i] = o;
        local += o * o;
    }

    red[tid] = local;
    __syncthreads();
    for (int s = 128; s > 0; s >>= 1) {
        if (tid < s) red[tid] += red[tid + s];
        __syncthreads();
    }
    if (tid == 0) partials[blockIdx.x] = red[0];
}

// 1024 partials -> one norm^2 scalar (deterministic fixed-order reduction).
__global__ void qaoa_reduce(const float* __restrict__ partials,
                            float* __restrict__ out)
{
    __shared__ float red[256];
    const int tid = threadIdx.x;
    red[tid] = partials[tid] + partials[tid + 256] +
               partials[tid + 512] + partials[tid + 768];
    __syncthreads();
    for (int s = 128; s > 0; s >>= 1) {
        if (tid < s) red[tid] += red[tid + s];
        __syncthreads();
    }
    if (tid == 0) out[0] = red[0];
}

// probabilities = amps^2 / norm^2 (in place over the probabilities region).
__global__ void qaoa_prob(float* __restrict__ p,
                          const float* __restrict__ normsq)
{
    const int i = blockIdx.x * blockDim.x + threadIdx.x;
    const float inv = 1.0f / normsq[0];
    const float v = p[i];
    p[i] = v * v * inv;
}

// ---------------------------------------------------------------------------
// Top-10 + final outputs. Single block of 256 threads: per-thread top-10
// insertion list over a 4096-element strided range, then LDS merge tree
// (tie rule: equal value -> lower index, matching jax.lax.top_k), then
// recompute the 10 costs exactly in f32 and argmin.
// ---------------------------------------------------------------------------
__global__ void qaoa_topk(const float* __restrict__ p,
                          const float* __restrict__ con,
                          float* __restrict__ opt_assign,
                          float* __restrict__ opt_cost,
                          float* __restrict__ top_assign,
                          float* __restrict__ assign_costs)
{
    __shared__ float sv[256 * TOPK];
    __shared__ int   si[256 * TOPK];
    __shared__ float cbuf[TOPK];
    const int tid = threadIdx.x;

    float tv[TOPK];
    int   ti[TOPK];
    #pragma unroll
    for (int r = 0; r < TOPK; ++r) { tv[r] = -1.0f; ti[r] = 0x7fffffff; }

    for (int i = tid; i < NSTATES; i += 256) {
        const float v = p[i];
        if (v > tv[TOPK - 1] || (v == tv[TOPK - 1] && i < ti[TOPK - 1])) {
            int r = TOPK - 1;
            while (r > 0 && (v > tv[r - 1] ||
                             (v == tv[r - 1] && i < ti[r - 1]))) {
                tv[r] = tv[r - 1]; ti[r] = ti[r - 1]; --r;
            }
            tv[r] = v; ti[r] = i;
        }
    }
    for (int r = 0; r < TOPK; ++r) {
        sv[tid * TOPK + r] = tv[r];
        si[tid * TOPK + r] = ti[r];
    }
    __syncthreads();

    for (int s = 128; s >= 1; s >>= 1) {
        if (tid < s) {
            float av[TOPK], bv[TOPK];
            int   ai[TOPK], bi[TOPK];
            for (int r = 0; r < TOPK; ++r) {
                av[r] = sv[tid * TOPK + r];        ai[r] = si[tid * TOPK + r];
                bv[r] = sv[(tid + s) * TOPK + r];  bi[r] = si[(tid + s) * TOPK + r];
            }
            int pa = 0, pb = 0;
            for (int r = 0; r < TOPK; ++r) {
                bool takeA;
                if (pa >= TOPK)      takeA = false;
                else if (pb >= TOPK) takeA = true;
                else takeA = (av[pa] > bv[pb]) ||
                             (av[pa] == bv[pb] && ai[pa] <= bi[pb]);
                if (takeA) { sv[tid * TOPK + r] = av[pa]; si[tid * TOPK + r] = ai[pa]; ++pa; }
                else       { sv[tid * TOPK + r] = bv[pb]; si[tid * TOPK + r] = bi[pb]; ++pb; }
            }
        }
        __syncthreads();
    }

    if (tid < TOPK) {
        const int idx = si[tid];
        float pen = 0.0f;
        for (int c = 0; c < NCON; ++c) {
            float viol = 0.0f;
            for (int k = 0; k < NVAR; ++k)
                viol += (float)((idx >> (NVAR - 1 - k)) & 1) * con[c * NVAR + k];
            float x = fmaxf(viol - 1.0f, 0.0f);
            pen += x * x;
        }
        const float cost = pen + 0.1f * (float)__popc(idx);
        for (int k = 0; k < NVAR; ++k)
            top_assign[tid * NVAR + k] = (float)((idx >> (NVAR - 1 - k)) & 1);
        assign_costs[tid] = cost;
        cbuf[tid] = cost;
    }
    __syncthreads();

    if (tid == 0) {
        int best = 0;
        for (int r = 1; r < TOPK; ++r)
            if (cbuf[r] < cbuf[best]) best = r;   // first min, as jnp.argmin
        const int idx = si[best];
        for (int k = 0; k < NVAR; ++k)
            opt_assign[k] = (float)((idx >> (NVAR - 1 - k)) & 1);
        opt_cost[0] = cbuf[best];
    }
}

// ---------------------------------------------------------------------------
extern "C" void kernel_launch(void* const* d_in, const int* in_sizes, int n_in,
                              void* d_out, int out_size, void* d_ws, size_t ws_size,
                              hipStream_t stream)
{
    const float* con   = (const float*)d_in[0];   // [64, 20]
    const float* beta  = (const float*)d_in[1];   // [3]
    const float* gamma = (const float*)d_in[2];   // [3]

    float* out          = (float*)d_out;
    float* opt_assign   = out;                     // 20
    float* opt_cost     = out + NVAR;              // 1
    float* probs        = out + NVAR + 1;          // 2^20 (doubles as amps buf)
    float* top_assign   = probs + NSTATES;         // 10*20
    float* assign_costs = top_assign + TOPK * NVAR;// 10

    char*  w        = (char*)d_ws;
    float* costs    = (float*)w;                               // 4 MB
    float* bbuf     = (float*)(w + (size_t)NSTATES * 4);       // 4 MB
    float* partials = (float*)(w + (size_t)2 * NSTATES * 4);   // 1024 floats
    float* normsq   = partials + 1024;                         // 3 floats

    qaoa_cost_wmma<<<NSTATES / 256, 32, 0, stream>>>(con, costs);

    for (int layer = 0; layer < NLAYER; ++layer) {
        qaoa_scale_cos<<<NSTATES / 256, 256, 0, stream>>>(
            probs, costs, gamma, layer,
            (layer == 0) ? normsq : (normsq + layer - 1), bbuf);
        qaoa_mix<<<NSTATES / 1024, 256, 0, stream>>>(bbuf, probs, beta, layer,
                                                     partials);
        qaoa_reduce<<<1, 256, 0, stream>>>(partials, normsq + layer);
    }

    qaoa_prob<<<NSTATES / 256, 256, 0, stream>>>(probs, normsq + 2);
    qaoa_topk<<<1, 256, 0, stream>>>(probs, con, opt_assign, opt_cost,
                                     top_assign, assign_costs);
}
